// lstm_source_79980880986534
// MI455X (gfx1250) — matrix operand order repaired and checked by reading, not compiled, not run on hardware
//
#include <hip/hip_runtime.h>
#include <hip/hip_bf16.h>

// ---------------------------------------------------------------------------
// 4-layer LSTM (relu activation) + dense head for MI455X (gfx1250, wave32).
//   - bf16 WMMA GEMMs (f32 accum) for input projections / dense head.
//   - Fused per-timestep recurrent kernel: each block owns a 16-wide hidden
//     slice and computes all 4 gates for it, then does the c/h update.
//   - Global->LDS staging uses CDNA5 async-tensor path:
//     global_load_async_to_lds_b128 + s_wait_asynccnt, double-buffered so
//     HBM/L2 traffic overlaps the v_wmma stream.
// ---------------------------------------------------------------------------

typedef unsigned short u16;
typedef __attribute__((ext_vector_type(8)))  __bf16 v8bf;
typedef __attribute__((ext_vector_type(16))) __bf16 v16bf;
typedef __attribute__((ext_vector_type(8)))  float  v8f;

#define LSTM_B  32
#define LSTM_T  256
#define LSTM_H  1024
#define LSTM_G  4096   // 4*H
#define LSTM_L  4

#define LDT 72   // padded LDS row stride (bf16 elems); 144B row, 16B aligned

__device__ __forceinline__ u16 f2bf(float f) {
  unsigned int u = __float_as_uint(f);
  unsigned int r = (u >> 16) & 1u;
  u += 0x7fffu + r;               // round-to-nearest-even
  return (u16)(u >> 16);
}

__device__ __forceinline__ float sigmoidf_(float x) {
  return 1.0f / (1.0f + __expf(-x));
}

// CDNA5 async copy: global memory -> LDS, 128b per lane, tracked by ASYNCcnt.
__device__ __forceinline__ void async_b128(unsigned lds_byte, const void* gaddr) {
  asm volatile("global_load_async_to_lds_b128 %0, %1, off"
               :: "v"(lds_byte), "v"(gaddr) : "memory");
}
__device__ __forceinline__ void wait_async0() {
  asm volatile("s_wait_asynccnt 0x0" ::: "memory");
}
__device__ __forceinline__ void wait_async3() {
  asm volatile("s_wait_asynccnt 0x3" ::: "memory");
}
__device__ __forceinline__ void wait_async6() {
  asm volatile("s_wait_asynccnt 0x6" ::: "memory");
}
__device__ __forceinline__ unsigned lds_addr(const void* p) {
  return (unsigned)(unsigned long long)p;  // low 32 bits of LDS generic addr
}

// ----------------------- conversion / init kernels --------------------------

__global__ void k_convert_wT(const float* __restrict__ src, u16* __restrict__ dst,
                             int K, int N, long total) {
  long idx = (long)blockIdx.x * blockDim.x + threadIdx.x;
  if (idx >= total) return;
  long nk = (long)N * K;
  long l  = idx / nk;
  long r  = idx - l * nk;
  int  n  = (int)(r / K);
  int  k  = (int)(r - (long)n * K);
  dst[idx] = f2bf(src[l * nk + (long)k * N + n]);
}

__global__ void k_convert_act(const float* __restrict__ src, u16* __restrict__ dst,
                              long total) {
  long idx = (long)blockIdx.x * blockDim.x + threadIdx.x;
  if (idx >= total) return;
  dst[idx] = f2bf(src[idx]);
}

__global__ void k_init_state0(const float* __restrict__ h0, const float* __restrict__ c0,
                              float* __restrict__ h, u16* __restrict__ hb,
                              float* __restrict__ c, int total) {
  int idx = blockIdx.x * blockDim.x + threadIdx.x;
  if (idx >= total) return;
  float hv = h0[idx];
  h[idx]  = hv;
  hb[idx] = f2bf(hv);
  c[idx]  = c0[idx];
}

__global__ void k_zero_c(float* __restrict__ c, int total) {
  int idx = blockIdx.x * blockDim.x + threadIdx.x;
  if (idx < total) c[idx] = 0.0f;
}

__global__ void k_copy_f32(const float* __restrict__ src, float* __restrict__ dst,
                           int total) {
  int idx = blockIdx.x * blockDim.x + threadIdx.x;
  if (idx < total) dst[idx] = src[idx];
}

// ----------------------- WMMA GEMM (bf16 in, f32 accum) --------------------
// C[M,N] = A[M,K] * Bt[N,K]^T + bias[N]
// Block: 256 thr = 8 waves (2 M-waves x 4 N-waves); block tile 64(M) x 128(N);
// each wave computes a 32x32 output (2x2 WMMA tiles). K staged in 64-wide
// chunks via double-buffered async LDS copies -> 8 WMMAs per barrier pair.

__global__ void k_gemm_wmma(const u16* __restrict__ A, const u16* __restrict__ Bt,
                            const float* __restrict__ bias, float* __restrict__ C,
                            int M, int N, int K, int scatter) {
  __shared__ u16 Als[2][64 * LDT];
  __shared__ u16 Bls[2][128 * LDT];

  const int tid  = threadIdx.x;
  const int lane = tid & 31;
  const int w    = tid >> 5;
  const int wm   = w & 1;        // M half (0..1)
  const int wn   = w >> 1;       // N quarter (0..3)
  const int mt0 = blockIdx.y * 64;
  const int nt0 = blockIdx.x * 128;

  v8f acc[2][2] = {};

  // staging geometry: A tile 64x64 (16 elems/thread), B tile 128x64 (32/thread)
  const int a_row = tid >> 2, a_col = (tid & 3) * 16;
  const int b_row = tid >> 1, b_col = (tid & 1) * 32;
  const u16* Ag = &A[(size_t)(mt0 + a_row) * K + a_col];
  const u16* Bg = &Bt[(size_t)(nt0 + b_row) * K + b_col];

  const int nch = K >> 6;
  auto issue = [&](int buf, int kk) {
    unsigned ao = lds_addr(&Als[buf][a_row * LDT + a_col]);
    async_b128(ao,      Ag + kk);
    async_b128(ao + 16, Ag + kk + 8);
    unsigned bo = lds_addr(&Bls[buf][b_row * LDT + b_col]);
    async_b128(bo,      Bg + kk);
    async_b128(bo + 16, Bg + kk + 8);
    async_b128(bo + 32, Bg + kk + 16);
    async_b128(bo + 48, Bg + kk + 24);
  };

  issue(0, 0);
  for (int c = 0; c < nch; ++c) {
    const int cur = c & 1;
    const bool more = (c + 1) < nch;
    if (more) { issue(1 - cur, (c + 1) << 6); wait_async6(); }
    else      { wait_async0(); }
    __syncthreads();

    const u16* As = Als[cur];
    const u16* Bs = Bls[cur];
#pragma unroll
    for (int ks = 0; ks < 64; ks += 32) {
      const int kb = ((lane >> 4) << 3) + ks;   // A frag: K chunks at kb, kb+16
      const int kh = ((lane >> 4) << 4) + ks;   // B frag: K chunks at kb, kb+16 (same as A)
      v16bf av[2], bv[2];
#pragma unroll
      for (int i = 0; i < 2; ++i) {
        const int ar = wm * 32 + i * 16 + (lane & 15);
        v8bf a0 = *(const v8bf*)&As[ar * LDT + kb];
        v8bf a1 = *(const v8bf*)&As[ar * LDT + kb + 16];
        av[i] = __builtin_shufflevector(a0, a1, 0,1,2,3,4,5,6,7,8,9,10,11,12,13,14,15);
      }
#pragma unroll
      for (int j = 0; j < 2; ++j) {
        const int br = wn * 32 + j * 16 + (lane & 15);
        v8bf b0 = *(const v8bf*)&Bs[br * LDT + kb];
        v8bf b1 = *(const v8bf*)&Bs[br * LDT + kb + 16];
        bv[j] = __builtin_shufflevector(b0, b1, 0,1,2,3,4,5,6,7,8,9,10,11,12,13,14,15);
      }
#pragma unroll
      for (int i = 0; i < 2; ++i)
#pragma unroll
        for (int j = 0; j < 2; ++j)
          acc[i][j] = __builtin_amdgcn_wmma_f32_16x16x32_bf16(
              false, av[i], false, bv[j], (short)0, acc[i][j], false, false);
    }
    __syncthreads();
  }

  const int mhi = (lane >> 4) << 3;
#pragma unroll
  for (int i = 0; i < 2; ++i) {
#pragma unroll
    for (int j2 = 0; j2 < 2; ++j2) {
      const int n = nt0 + wn * 32 + j2 * 16 + (lane & 15);
      const float bb = bias ? bias[n] : 0.0f;
#pragma unroll
      for (int j = 0; j < 8; ++j) {
        int m = mt0 + wm * 32 + i * 16 + mhi + j;
        float v = acc[i][j2][j] + bb;
        size_t o;
        if (scatter) {
          int b = m >> 8;          // m / T (T=256)
          int t = m & 255;         // m % T
          o = ((size_t)t * LSTM_B + b) * (size_t)N + n;
        } else {
          o = (size_t)m * N + n;
        }
        C[o] = v;
      }
    }
  }
}

// ----------------------- fused LSTM step -----------------------------------
// One timestep. grid = H/16 blocks; block = 256 thr = 8 waves.
// Wave w: mtile=w&1 (batch rows), gate=w>>1 (i,f,g,o). Block owns hidden
// slice [j0, j0+16): WMMA z = h@U for all 4 gates, then gate math in-block.
// K staged in 64-wide chunks, double-buffered async LDS copies.

__global__ void k_lstm_step(int t,
                            const u16* __restrict__ hbin, u16* __restrict__ hbout,
                            float* __restrict__ hfull, float* __restrict__ cbuf,
                            const u16* __restrict__ Ut,     // [4H][H] bf16
                            const float* __restrict__ xz,   // [T][B][4H] f32
                            float* __restrict__ outb) {     // [B][T][H] f32
  __shared__ u16  Als[2][32 * LDT];
  __shared__ u16  Bls[2][64 * LDT];
  __shared__ float zs[4][32][17];

  const int tid  = threadIdx.x;
  const int lane = tid & 31;
  const int w    = tid >> 5;
  const int mtile = w & 1;
  const int gi    = w >> 1;          // gate index 0..3
  const int j0    = blockIdx.x * 16; // hidden slice base

  v8f acc = {};

  // staging: A tile 32x64 (8 elems/thread), B tile 64x64 (16 elems/thread)
  const int a_row = tid >> 3, a_col = (tid & 7) * 8;
  const int b_row = tid >> 2, b_col = (tid & 3) * 16;
  const int b_gate = b_row >> 4, b_j = b_row & 15;
  const u16* Ag = &hbin[(size_t)a_row * LSTM_H + a_col];
  const u16* Bg = &Ut[((size_t)b_gate * LSTM_H + j0 + b_j) * LSTM_H + b_col];

  auto issue = [&](int buf, int kk) {
    async_b128(lds_addr(&Als[buf][a_row * LDT + a_col]), Ag + kk);
    unsigned bo = lds_addr(&Bls[buf][b_row * LDT + b_col]);
    async_b128(bo,      Bg + kk);
    async_b128(bo + 16, Bg + kk + 8);
  };

  const int nch = LSTM_H >> 6;       // 16 chunks
  issue(0, 0);
  for (int c = 0; c < nch; ++c) {
    const int cur = c & 1;
    const bool more = (c + 1) < nch;
    if (more) { issue(1 - cur, (c + 1) << 6); wait_async3(); }
    else      { wait_async0(); }
    __syncthreads();

    const u16* As = Als[cur];
    const u16* Bs = Bls[cur];
#pragma unroll
    for (int ks = 0; ks < 64; ks += 32) {
      const int kb = ((lane >> 4) << 3) + ks;
      const int kh = ((lane >> 4) << 4) + ks;
      const int ar = mtile * 16 + (lane & 15);
      const int br = gi * 16 + (lane & 15);   // B rows grouped by gate

      v8bf a0 = *(const v8bf*)&As[ar * LDT + kb];
      v8bf a1 = *(const v8bf*)&As[ar * LDT + kb + 16];
      v16bf av = __builtin_shufflevector(a0, a1, 0,1,2,3,4,5,6,7,8,9,10,11,12,13,14,15);

      v8bf b0 = *(const v8bf*)&Bs[br * LDT + kb];
      v8bf b1 = *(const v8bf*)&Bs[br * LDT + kb + 16];
      v16bf bv = __builtin_shufflevector(b0, b1, 0,1,2,3,4,5,6,7,8,9,10,11,12,13,14,15);

      acc = __builtin_amdgcn_wmma_f32_16x16x32_bf16(false, av, false, bv,
                                                    (short)0, acc, false, false);
    }
    __syncthreads();
  }

  // stash recurrent contribution: zs[gate][batch][hidden-in-slice]
  const int mhi = (lane >> 4) << 3;
#pragma unroll
  for (int j = 0; j < 8; ++j) {
    zs[gi][mtile * 16 + mhi + j][lane & 15] = acc[j];
  }
  __syncthreads();

  // gate math + state update: 32 batches x 16 hidden units = 512 items
  for (int p = tid; p < 512; p += 256) {
    const int m  = p >> 4;
    const int jj = p & 15;
    const int j  = j0 + jj;
    const size_t xb = ((size_t)t * LSTM_B + m) * (size_t)LSTM_G;
    float zi = zs[0][m][jj] + xz[xb + 0 * LSTM_H + j];
    float zf = zs[1][m][jj] + xz[xb + 1 * LSTM_H + j];
    float zg = zs[2][m][jj] + xz[xb + 2 * LSTM_H + j];
    float zo = zs[3][m][jj] + xz[xb + 3 * LSTM_H + j];
    float ig = sigmoidf_(zi);
    float fg = sigmoidf_(zf);
    float gg = zg > 0.0f ? zg : 0.0f;          // activation='relu'
    float og = sigmoidf_(zo);
    const int ci = m * LSTM_H + j;
    float cn = fg * cbuf[ci] + ig * gg;
    cbuf[ci] = cn;
    float hh = og * (cn > 0.0f ? cn : 0.0f);   // relu on cell output
    hfull[ci]  = hh;
    hbout[ci]  = f2bf(hh);
    outb[((size_t)m * LSTM_T + t) * LSTM_H + j] = hh;
  }
}

// ----------------------- launch ---------------------------------------------

extern "C" void kernel_launch(void* const* d_in, const int* in_sizes, int n_in,
                              void* d_out, int out_size, void* d_ws, size_t ws_size,
                              hipStream_t stream) {
  (void)in_sizes; (void)n_in; (void)out_size; (void)ws_size;
  const float* x  = (const float*)d_in[0];   // [B,T,E]
  const float* h0 = (const float*)d_in[1];   // [B,H]
  const float* c0 = (const float*)d_in[2];   // [B,H]
  const float* W  = (const float*)d_in[3];   // [L,E,4H]
  const float* U  = (const float*)d_in[4];   // [L,H,4H]
  const float* bb = (const float*)d_in[5];   // [L,4H]
  const float* Wd = (const float*)d_in[6];   // [H,H]
  const float* bd = (const float*)d_in[7];   // [H]
  float* out = (float*)d_out;                // [B,T,H] ++ [B,H]

  char* ws = (char*)d_ws;
  const size_t szWt   = (size_t)LSTM_L * LSTM_G * LSTM_H * 2;
  const size_t szUt   = szWt;
  const size_t szWdt  = (size_t)LSTM_H * LSTM_H * 2;
  const size_t szXz   = (size_t)LSTM_T * LSTM_B * LSTM_G * 4;
  const size_t szAct  = (size_t)LSTM_B * LSTM_T * LSTM_H * 2;
  const size_t szOutB = (size_t)LSTM_B * LSTM_T * LSTM_H * 4;
  const size_t szH    = (size_t)LSTM_B * LSTM_H * 4;
  const size_t szHb   = (size_t)LSTM_B * LSTM_H * 2;

  size_t off = 0;
  u16*   Wt   = (u16*)(ws + off);   off += szWt;
  u16*   Ut   = (u16*)(ws + off);   off += szUt;
  u16*   Wdt  = (u16*)(ws + off);   off += szWdt;
  float* xz   = (float*)(ws + off); off += szXz;
  u16*   actb = (u16*)(ws + off);   off += szAct;
  float* outb = (float*)(ws + off); off += szOutB;
  float* hbuf = (float*)(ws + off); off += szH;
  float* cbuf = (float*)(ws + off); off += szH;
  u16*   hb0  = (u16*)(ws + off);   off += szHb;
  u16*   hb1  = (u16*)(ws + off);   off += szHb;

  const long nW  = (long)LSTM_L * LSTM_G * LSTM_H;
  const long nWd = (long)LSTM_H * LSTM_H;
  const long nAct = (long)LSTM_B * LSTM_T * LSTM_H;
  const int  nState = LSTM_B * LSTM_H;

  k_convert_wT<<<dim3((nW + 255) / 256), 256, 0, stream>>>(W, Wt, LSTM_H, LSTM_G, nW);
  k_convert_wT<<<dim3((nW + 255) / 256), 256, 0, stream>>>(U, Ut, LSTM_H, LSTM_G, nW);
  k_convert_wT<<<dim3((nWd + 255) / 256), 256, 0, stream>>>(Wd, Wdt, LSTM_H, LSTM_H, nWd);

  for (int l = 0; l < LSTM_L; ++l) {
    const float* src = (l == 0) ? x : outb;
    k_convert_act<<<dim3((nAct + 255) / 256), 256, 0, stream>>>(src, actb, nAct);

    // xz = act @ W[l] + b[l]  -> scattered to [T][B][4H]
    k_gemm_wmma<<<dim3(LSTM_G / 128, (LSTM_B * LSTM_T) / 64), 256, 0, stream>>>(
        actb, Wt + (size_t)l * LSTM_G * LSTM_H, bb + (size_t)l * LSTM_G,
        xz, LSTM_B * LSTM_T, LSTM_G, LSTM_H, 1);

    if (l == 0) {
      k_init_state0<<<dim3(nState / 256), 256, 0, stream>>>(h0, c0, hbuf, hb0, cbuf, nState);
    } else {
      k_zero_c<<<dim3(nState / 256), 256, 0, stream>>>(cbuf, nState);
      // hbuf / hb0 already hold previous layer's final hidden state
    }

    for (int t = 0; t < LSTM_T; ++t) {
      const u16* hin = (t & 1) ? hb1 : hb0;
      u16*       hot = (t & 1) ? hb0 : hb1;
      k_lstm_step<<<dim3(LSTM_H / 16), 256, 0, stream>>>(
          t, hin, hot, hbuf, cbuf, Ut + (size_t)l * LSTM_G * LSTM_H, xz, outb);
    }
  }

  // dense head: out = outb @ Wd + bd  (row-major [B*T, H] == [B,T,H])
  k_convert_act<<<dim3((nAct + 255) / 256), 256, 0, stream>>>(outb, actb, nAct);
  k_gemm_wmma<<<dim3(LSTM_H / 128, (LSTM_B * LSTM_T) / 64), 256, 0, stream>>>(
      actb, Wdt, bd, out, LSTM_B * LSTM_T, LSTM_H, LSTM_H, 0);

  k_copy_f32<<<dim3(nState / 256), 256, 0, stream>>>(
      hbuf, out + (size_t)LSTM_B * LSTM_T * LSTM_H, nState);
}
